// Attention_63084479643850
// MI455X (gfx1250) — matrix operand order repaired and checked by reading, not compiled, not run on hardware
//
#include <hip/hip_runtime.h>
#include <hip/hip_bf16.h>

// ---------------------------------------------------------------------------
// Attention layer for MI455X (gfx1250, wave32, WMMA bf16 / f32-accumulate).
// Pipeline: convert f32->bf16, QKV WMMA GEMMs (double-buffered K=64 tiles),
// RoPE, flash attention (WMMA for QK^T and P*V with online softmax),
// output-projection WMMA GEMM.
// ---------------------------------------------------------------------------

typedef __bf16 bf16_t;
typedef __attribute__((ext_vector_type(8)))  __bf16 v8bf;
typedef __attribute__((ext_vector_type(16))) __bf16 v16bf;
typedef __attribute__((ext_vector_type(8)))  float  v8f;

union V8F { v8f v; float f[8]; };

#define S_LEN  2048
#define DMODEL 4096
#define NQH    32
#define NKVH   8
#define HDIM   128
#define KVDIM  (NKVH * HDIM)   // 1024

static __device__ inline v16bf cat8(v8bf lo, v8bf hi) {
    return __builtin_shufflevector(lo, hi, 0,1,2,3,4,5,6,7,8,9,10,11,12,13,14,15);
}

// ---------------------------------------------------------------------------
// f32 -> bf16 convert, 8 elements/thread (n must be a multiple of 8; all our
// tensors are multiples of 4096).
// ---------------------------------------------------------------------------
__global__ void f32_to_bf16_kernel(const float* __restrict__ in,
                                   bf16_t* __restrict__ out, int n8) {
    int i = blockIdx.x * blockDim.x + threadIdx.x;
    if (i < n8) {
        v8f v = *(const v8f*)(in + (size_t)i * 8);
        v8bf b;
        #pragma unroll
        for (int e = 0; e < 8; e++) b[e] = (bf16_t)v[e];
        *(v8bf*)(out + (size_t)i * 8) = b;
    }
}

// ---------------------------------------------------------------------------
// C[M,N] = A[M,K] * W[N,K]^T   (A,W bf16; C f32 or bf16)
// Block: 256 thr (8 waves). Block tile 128x128, K-tile 64, register
// double-buffering of the next tile overlapped with 16 WMMAs per stage.
// Wave (wm 0..3, wn 0..1) computes 32x64 via 2x4 WMMA accumulators.
// ---------------------------------------------------------------------------
template<bool OUT_BF16>
__global__ __launch_bounds__(256)
void gemm_bf16_kernel(const bf16_t* __restrict__ A,
                      const bf16_t* __restrict__ W,
                      void* __restrict__ Cout,
                      int M, int N, int K)
{
    __shared__ alignas(16) bf16_t lA[128][64];   // 16 KB
    __shared__ alignas(16) bf16_t lB[128][64];   // 16 KB, Bt layout: [n][k]

    const int tid  = threadIdx.x;
    const int lane = tid & 31;
    const int wid  = tid >> 5;
    const int wm   = wid & 3;        // row group of 32
    const int wn   = wid >> 2;       // col group of 64
    const int m0   = blockIdx.y * 128;
    const int n0   = blockIdx.x * 128;
    const int lrow = lane & 15;
    const int lhi  = lane >> 4;

    v8f acc[2][4];
    #pragma unroll
    for (int i = 0; i < 2; i++)
        #pragma unroll
        for (int j = 0; j < 4; j++) acc[i][j] = {};

    // Per-thread staging chunks: 128x64 bf16 tile = 1024 x 16B chunks,
    // 4 chunks per thread. c = tid*4+i -> row=c>>3, ch=c&7.
    int crow[4], cch[4];
    #pragma unroll
    for (int i = 0; i < 4; i++) { int c = tid * 4 + i; crow[i] = c >> 3; cch[i] = c & 7; }

    // Prologue: load tile 0 into registers.
    v8bf ra[4], rb[4];
    #pragma unroll
    for (int i = 0; i < 4; i++) {
        ra[i] = *(const v8bf*)(A + (size_t)(m0 + crow[i]) * K + cch[i] * 8);
        rb[i] = *(const v8bf*)(W + (size_t)(n0 + crow[i]) * K + cch[i] * 8);
    }

    const int nkt = K / 64;
    for (int kt = 0; kt < nkt; kt++) {
        // Commit current tile registers to LDS.
        #pragma unroll
        for (int i = 0; i < 4; i++) {
            *(v8bf*)&lA[crow[i]][cch[i] * 8] = ra[i];
            *(v8bf*)&lB[crow[i]][cch[i] * 8] = rb[i];
        }
        __syncthreads();

        // Kick off next tile's global loads; they retire while we do WMMAs.
        if (kt + 1 < nkt) {
            int k0n = (kt + 1) * 64;
            #pragma unroll
            for (int i = 0; i < 4; i++) {
                ra[i] = *(const v8bf*)(A + (size_t)(m0 + crow[i]) * K + k0n + cch[i] * 8);
                rb[i] = *(const v8bf*)(W + (size_t)(n0 + crow[i]) * K + k0n + cch[i] * 8);
            }
        }

        // 2 K-steps of 32 x (2 m-tiles x 4 n-tiles) = 16 WMMAs per stage.
        #pragma unroll
        for (int kk = 0; kk < 2; kk++) {
            const int kcol = kk * 32;
            v16bf af[2], bfr[4];
            #pragma unroll
            for (int mt = 0; mt < 2; mt++) {
                int m  = wm * 32 + mt * 16 + lrow;
                int kb = kcol + lhi * 8;
                af[mt] = cat8(*(const v8bf*)&lA[m][kb],
                              *(const v8bf*)&lA[m][16 + kb]);
            }
            #pragma unroll
            for (int nt = 0; nt < 4; nt++) {
                int n  = wn * 64 + nt * 16 + lrow;
                int kb = kcol + lhi * 8;
                bfr[nt] = cat8(*(const v8bf*)&lB[n][kb],
                               *(const v8bf*)&lB[n][kb + 16]);
            }
            #pragma unroll
            for (int mt = 0; mt < 2; mt++)
                #pragma unroll
                for (int nt = 0; nt < 4; nt++)
                    acc[mt][nt] = __builtin_amdgcn_wmma_f32_16x16x32_bf16(
                        false, af[mt], false, bfr[nt], (short)0, acc[mt][nt],
                        false, false);
        }
        __syncthreads();
    }

    // Epilogue: D layout lane -> n = l&15, reg r -> m = r + 8*(l>>4)
    #pragma unroll
    for (int mt = 0; mt < 2; mt++) {
        #pragma unroll
        for (int nt = 0; nt < 4; nt++) {
            V8F u; u.v = acc[mt][nt];
            int n = n0 + wn * 64 + nt * 16 + lrow;
            #pragma unroll
            for (int r = 0; r < 8; r++) {
                int m = m0 + wm * 32 + mt * 16 + r + lhi * 8;
                if constexpr (OUT_BF16)
                    ((bf16_t*)Cout)[(size_t)m * N + n] = (bf16_t)u.f[r];
                else
                    ((float*)Cout)[(size_t)m * N + n] = u.f[r];
            }
        }
    }
}

// ---------------------------------------------------------------------------
// RoPE, interleaved pairing (matches reference complex multiply).
// Reads f32 projection, writes bf16 for the attention WMMAs.
// ---------------------------------------------------------------------------
__global__ void rope_kernel(const float* __restrict__ Xf,
                            const float* __restrict__ freqs,
                            bf16_t* __restrict__ Xb, int nheads)
{
    int idx   = blockIdx.x * blockDim.x + threadIdx.x;
    int total = S_LEN * nheads * (HDIM / 2);
    if (idx >= total) return;
    int j = idx % (HDIM / 2);
    int h = (idx / (HDIM / 2)) % nheads;
    int s = idx / ((HDIM / 2) * nheads);
    float f  = freqs[s * (HDIM / 2) + j];
    float c  = __cosf(f);
    float sn = __sinf(f);
    size_t base = (size_t)s * (nheads * HDIM) + (size_t)h * HDIM + 2 * j;
    float t0 = Xf[base], t1 = Xf[base + 1];
    Xb[base]     = (bf16_t)(t0 * c - t1 * sn);
    Xb[base + 1] = (bf16_t)(t0 * sn + t1 * c);
}

// ---------------------------------------------------------------------------
// Flash attention: grid (head, qtile of 64). Block 128 thr = 4 waves,
// each wave owns 16 query rows. KV tiles of 32 keys staged in LDS.
// GQA: head h uses kv head h/4 (repeat_interleave semantics).
// ---------------------------------------------------------------------------
__global__ __launch_bounds__(128)
void attn_kernel(const bf16_t* __restrict__ Q,   // [S, 4096] head-major
                 const bf16_t* __restrict__ Kb,  // [S, 1024]
                 const bf16_t* __restrict__ Vb,  // [S, 1024]
                 bf16_t* __restrict__ Ob)        // [S, 4096]
{
    __shared__ alignas(16) bf16_t lK [32][HDIM];   // key rows (Bt for QK^T)
    __shared__ alignas(16) bf16_t lVt[HDIM][32];   // V transposed (Bt for P*V)
    __shared__ alignas(16) bf16_t lP [4][16][32];  // per-wave P redistribute

    const int h    = blockIdx.x;
    const int q0   = blockIdx.y * 64;
    const int kvh  = h >> 2;                // 32 q-heads -> 8 kv-heads
    const int tid  = threadIdx.x;
    const int lane = tid & 31;
    const int wid  = tid >> 5;              // wave: query rows q0+wid*16..
    const int lrow = lane & 15;
    const int lhi  = lane >> 4;
    const float scale = 0.08838834764831845f;  // 1/sqrt(128)

    // Q fragments for all 4 K-steps of the 128-wide head dim, kept in VGPRs.
    v16bf qf[4];
    {
        const bf16_t* qrow =
            Q + (size_t)(q0 + wid * 16 + lrow) * DMODEL + (size_t)h * HDIM;
        int kb = lhi * 8;
        #pragma unroll
        for (int ks = 0; ks < 4; ks++)
            qf[ks] = cat8(*(const v8bf*)(qrow + ks * 32 + kb),
                          *(const v8bf*)(qrow + ks * 32 + 16 + kb));
    }

    v8f oacc[8];
    #pragma unroll
    for (int i = 0; i < 8; i++) oacc[i] = {};
    float rowM[8], rowL[8];
    #pragma unroll
    for (int r = 0; r < 8; r++) { rowM[r] = -1e30f; rowL[r] = 0.f; }

    for (int k0 = 0; k0 < S_LEN; k0 += 32) {
        // Stage K tile (32x128 bf16): 512 x 16B chunks, 4 per thread.
        #pragma unroll
        for (int i = 0; i < 4; i++) {
            int c = tid * 4 + i;
            int row = c >> 4, ch = c & 15;
            *(v8bf*)&lK[row][ch * 8] =
                *(const v8bf*)(Kb + (size_t)(k0 + row) * KVDIM +
                               (size_t)kvh * HDIM + ch * 8);
        }
        // Stage V transposed: thread handles key = tid&31, d-quarter = tid>>5.
        {
            int key = tid & 31;
            int dq  = (tid >> 5) * 32;
            const bf16_t* vrow = Vb + (size_t)(k0 + key) * KVDIM +
                                 (size_t)kvh * HDIM + dq;
            #pragma unroll
            for (int cc = 0; cc < 4; cc++) {
                v8bf vv = *(const v8bf*)(vrow + cc * 8);
                #pragma unroll
                for (int e = 0; e < 8; e++)
                    lVt[dq + cc * 8 + e][key] = vv[e];
            }
        }
        __syncthreads();

        // Scores S = q * K^T : 16x32 per wave, via 4 K-steps x 2 N-tiles.
        v8f sa0 = {}, sa1 = {};
        #pragma unroll
        for (int ks = 0; ks < 4; ks++) {
            {
                int key = lrow, d = ks * 32 + lhi * 8;
                v16bf kf = cat8(*(const v8bf*)&lK[key][d],
                                *(const v8bf*)&lK[key][d + 16]);
                sa0 = __builtin_amdgcn_wmma_f32_16x16x32_bf16(
                    false, qf[ks], false, kf, (short)0, sa0, false, false);
            }
            {
                int key = 16 + lrow, d = ks * 32 + lhi * 8;
                v16bf kf = cat8(*(const v8bf*)&lK[key][d],
                                *(const v8bf*)&lK[key][d + 16]);
                sa1 = __builtin_amdgcn_wmma_f32_16x16x32_bf16(
                    false, qf[ks], false, kf, (short)0, sa1, false, false);
            }
        }

        // Online softmax over this 32-key tile.
        // Row stats live per-lane-half: reg r <-> row m = r + 8*(lane>=16).
        V8F s0, s1; s0.v = sa0; s1.v = sa1;
        float alpha[8];
        #pragma unroll
        for (int r = 0; r < 8; r++) {
            float x0 = s0.f[r] * scale, x1 = s1.f[r] * scale;
            float mx = fmaxf(x0, x1);
            #pragma unroll
            for (int off = 1; off < 16; off <<= 1)
                mx = fmaxf(mx, __shfl_xor(mx, off, 32));
            float newM = fmaxf(rowM[r], mx);
            float p0 = __expf(x0 - newM), p1 = __expf(x1 - newM);
            float ps = p0 + p1;
            #pragma unroll
            for (int off = 1; off < 16; off <<= 1)
                ps += __shfl_xor(ps, off, 32);
            float al = __expf(rowM[r] - newM);
            rowL[r]  = rowL[r] * al + ps;
            rowM[r]  = newM;
            alpha[r] = al;
            // Redistribute P into A-fragment layout through per-wave LDS.
            int m = r + lhi * 8;
            lP[wid][m][lrow]      = (bf16_t)p0;
            lP[wid][m][lrow + 16] = (bf16_t)p1;
        }
        V8F av;
        #pragma unroll
        for (int r = 0; r < 8; r++) av.f[r] = alpha[r];
        #pragma unroll
        for (int i = 0; i < 8; i++) oacc[i] *= av.v;

        // P(16x32) * V(32x128) accumulated into O. LDS ops are in-order per
        // wave; only this wave touches lP[wid], so no barrier needed here.
        v16bf pf;
        {
            int kb = lhi * 8;
            pf = cat8(*(const v8bf*)&lP[wid][lrow][kb],
                      *(const v8bf*)&lP[wid][lrow][16 + kb]);
        }
        #pragma unroll
        for (int dt = 0; dt < 8; dt++) {
            int d  = dt * 16 + lrow;
            int kk = lhi * 8;
            v16bf vf = cat8(*(const v8bf*)&lVt[d][kk],
                            *(const v8bf*)&lVt[d][kk + 16]);
            oacc[dt] = __builtin_amdgcn_wmma_f32_16x16x32_bf16(
                false, pf, false, vf, (short)0, oacc[dt], false, false);
        }
        __syncthreads();
    }

    // Finalize: divide by row sums, write bf16 head-major output.
    float rinv[8];
    #pragma unroll
    for (int r = 0; r < 8; r++) rinv[r] = 1.0f / rowL[r];
    #pragma unroll
    for (int dt = 0; dt < 8; dt++) {
        V8F u; u.v = oacc[dt];
        int d = h * HDIM + dt * 16 + lrow;
        #pragma unroll
        for (int r = 0; r < 8; r++) {
            int m = q0 + wid * 16 + r + lhi * 8;
            Ob[(size_t)m * DMODEL + d] = (bf16_t)(u.f[r] * rinv[r]);
        }
    }
}

// ---------------------------------------------------------------------------
extern "C" void kernel_launch(void* const* d_in, const int* in_sizes, int n_in,
                              void* d_out, int out_size, void* d_ws, size_t ws_size,
                              hipStream_t stream)
{
    const float* x  = (const float*)d_in[0];
    const float* fr = (const float*)d_in[1];
    const float* wq = (const float*)d_in[2];
    const float* wk = (const float*)d_in[3];
    const float* wv = (const float*)d_in[4];
    const float* wo = (const float*)d_in[5];

    char* ws = (char*)d_ws;
    size_t off = 0;
    auto alloc = [&](size_t bytes) {
        void* p = ws + off;
        off = (off + bytes + 255) & ~(size_t)255;
        return p;
    };

    bf16_t* xb  = (bf16_t*)alloc((size_t)S_LEN * DMODEL * 2);
    bf16_t* wqb = (bf16_t*)alloc((size_t)DMODEL * DMODEL * 2);
    bf16_t* wkb = (bf16_t*)alloc((size_t)KVDIM  * DMODEL * 2);
    bf16_t* wvb = (bf16_t*)alloc((size_t)KVDIM  * DMODEL * 2);
    bf16_t* wob = (bf16_t*)alloc((size_t)DMODEL * DMODEL * 2);
    float*  Qf  = (float*) alloc((size_t)S_LEN * DMODEL * 4);
    float*  Kf  = (float*) alloc((size_t)S_LEN * KVDIM  * 4);
    bf16_t* vb  = (bf16_t*)alloc((size_t)S_LEN * KVDIM  * 2);
    bf16_t* qb  = (bf16_t*)alloc((size_t)S_LEN * DMODEL * 2);
    bf16_t* kb  = (bf16_t*)alloc((size_t)S_LEN * KVDIM  * 2);
    bf16_t* ab  = (bf16_t*)alloc((size_t)S_LEN * DMODEL * 2);

    // 1) f32 -> bf16 conversions (8 elems/thread)
    {
        const float* srcs[5] = { x, wq, wk, wv, wo };
        bf16_t*      dsts[5] = { xb, wqb, wkb, wvb, wob };
        int          cnts[5] = { S_LEN * DMODEL, DMODEL * DMODEL,
                                 KVDIM * DMODEL, KVDIM * DMODEL,
                                 DMODEL * DMODEL };
        for (int i = 0; i < 5; i++) {
            int n8 = cnts[i] / 8;
            f32_to_bf16_kernel<<<(n8 + 255) / 256, 256, 0, stream>>>(
                srcs[i], dsts[i], n8);
        }
    }

    // 2) QKV projections
    gemm_bf16_kernel<false><<<dim3(DMODEL / 128, S_LEN / 128), 256, 0, stream>>>(
        xb, wqb, Qf, S_LEN, DMODEL, DMODEL);
    gemm_bf16_kernel<false><<<dim3(KVDIM / 128, S_LEN / 128), 256, 0, stream>>>(
        xb, wkb, Kf, S_LEN, KVDIM, DMODEL);
    gemm_bf16_kernel<true ><<<dim3(KVDIM / 128, S_LEN / 128), 256, 0, stream>>>(
        xb, wvb, vb, S_LEN, KVDIM, DMODEL);

    // 3) RoPE
    {
        int nq = S_LEN * NQH * (HDIM / 2);
        rope_kernel<<<(nq + 255) / 256, 256, 0, stream>>>(Qf, fr, qb, NQH);
        int nk = S_LEN * NKVH * (HDIM / 2);
        rope_kernel<<<(nk + 255) / 256, 256, 0, stream>>>(Kf, fr, kb, NKVH);
    }

    // 4) Attention
    attn_kernel<<<dim3(NQH, S_LEN / 64), 128, 0, stream>>>(qb, kb, vb, ab);

    // 5) Output projection -> f32 result
    gemm_bf16_kernel<false><<<dim3(DMODEL / 128, S_LEN / 128), 256, 0, stream>>>(
        ab, wob, (float*)d_out, S_LEN, DMODEL, DMODEL);
}
